// GroupQueryAttention_37108517437747
// MI455X (gfx1250) — matrix operand order repaired and checked by reading, not compiled, not run on hardware
//
#include <hip/hip_runtime.h>

// ---------------------------------------------------------------------------
// GQA forward for MI455X (gfx1250, wave32, WMMA + TDM async tensor loads).
//  d_out = [ out: B*S*D fp32 | scores(softmaxed): B*H*S*S fp32 ]
//  d_ws  = bf16 staging (~33 MB)
// ---------------------------------------------------------------------------

#define DEV __device__ __forceinline__

typedef __attribute__((ext_vector_type(16))) __bf16 v16bf;
typedef __attribute__((ext_vector_type(8)))  float  v8f;
typedef __attribute__((ext_vector_type(4)))  float  f32x4;
typedef __attribute__((ext_vector_type(4)))  unsigned int u32x4;
typedef __attribute__((ext_vector_type(8)))  int    i32x8;
typedef __attribute__((ext_vector_type(4)))  int    i32x4;

union Frag { v16bf v; uint4 u[2]; };

DEV unsigned short f2bf(float f) {
  unsigned u = __builtin_bit_cast(unsigned, f);
  u += 0x7FFFu + ((u >> 16) & 1u);          // round-to-nearest-even
  return (unsigned short)(u >> 16);
}

DEV v8f wmma_bf16(Frag a, Frag b, v8f c) {
  return __builtin_amdgcn_wmma_f32_16x16x32_bf16(
      /*neg_a=*/false, a.v, /*neg_b=*/false, b.v,
      /*c_mod=*/(short)0, c, /*reuse_a=*/false, /*reuse_b=*/false);
}

// Wave-uniform "am I wave 0 of this block" — readfirstlane pins the wave id
// in an SGPR so the guard lowers to a scalar branch (TDM ignores EXEC, so we
// must actually *branch*, not predicate, to issue the DMA from one wave only).
DEV bool is_wave0() {
  return __builtin_amdgcn_readfirstlane((int)(threadIdx.x >> 5)) == 0;
}

// problem dims (fixed by the reference)
constexpr int BB  = 2;
constexpr int SEQ = 2048;
constexpr int DM  = 1024;
constexpr int NH  = 16;
constexpr int NG  = 4;
constexpr int HD  = 64;
constexpr int KVD = NG * HD;               // 256
constexpr float ATT_SCALE = 0.125f;        // 1/sqrt(64)

// tiling
constexpr int BM = 128;   // block rows
constexpr int BN = 64;    // block cols
constexpr int BK = 32;    // k-step (one wmma K)
constexpr int LK = 40;    // padded LDS row stride (halfwords): 64B data + 16B pad

// ---------------------------------------------------------------------------
// TDM: async 2D bf16 tile load Global -> LDS with row padding matching LK.
// D# packing per CDNA5 ISA ch.8 (group0 128b, group1 256b; groups2/3 zero).
// pad: every 16 DWORDs (one 64B tile row) insert 4 DWORDs (16B) -> 80B rows.
// ---------------------------------------------------------------------------
DEV void tdm_load_2d_bf16(unsigned lds_off, const void* gptr,
                          unsigned rows, unsigned cols, unsigned stride_elems) {
  unsigned long long ga = (unsigned long long)(size_t)gptr;
  u32x4 g0;
  g0[0] = 1u;                                         // count=1, user mode
  g0[1] = lds_off;                                    // LDS byte address
  g0[2] = (unsigned)ga;                               // global_addr[31:0]
  g0[3] = (unsigned)((ga >> 32) & 0x01FFFFFFu) | (2u << 30);  // addr[56:32] | type=2
  i32x8 g1;
  g1[0] = (int)((1u << 16) |                          // data_size = 2 bytes
                (1u << 20) | (3u << 22) | (3u << 25));// pad_en, interval=16DW, amount=4DW
  g1[1] = (int)((cols & 0xFFFFu) << 16);              // tensor_dim0 = cols (lo16 @63:48)
  g1[2] = (int)((cols >> 16) | ((rows & 0xFFFFu) << 16)); // dim0 hi | tensor_dim1 lo
  g1[3] = (int)((rows >> 16) | (cols << 16));         // dim1 hi | tile_dim0 = cols
  g1[4] = (int)rows;                                  // tile_dim1 = rows, tile_dim2 = 0
  g1[5] = (int)stride_elems;                          // tensor_dim0_stride[31:0]
  g1[6] = 0;                                          // stride hi | dim1_stride lo
  g1[7] = 0;
  i32x4 z4 = {0, 0, 0, 0};
#if __clang_major__ >= 23
  i32x8 z8 = {0, 0, 0, 0, 0, 0, 0, 0};
  __builtin_amdgcn_tensor_load_to_lds(g0, g1, z4, z4, z8, 0);
#else
  __builtin_amdgcn_tensor_load_to_lds(g0, g1, z4, z4, 0);
#endif
}

// ---------------------------------------------------------------------------
// fp32 -> bf16 cast (vectorized x4)
// ---------------------------------------------------------------------------
__global__ __launch_bounds__(256)
void cast_f32_bf16(const float* __restrict__ in, unsigned short* __restrict__ out, int n4) {
  int i = blockIdx.x * 256 + threadIdx.x;
  if (i >= n4) return;
  f32x4 d = ((const f32x4*)in)[i];
  uint2 p;
  p.x = (unsigned)f2bf(d[0]) | ((unsigned)f2bf(d[1]) << 16);
  p.y = (unsigned)f2bf(d[2]) | ((unsigned)f2bf(d[3]) << 16);
  ((uint2*)out)[i] = p;
}

// ---------------------------------------------------------------------------
// LDS fragment helpers (ISA 16-bit A 16x32 / B 32x16 VGPR layouts)
// ---------------------------------------------------------------------------
DEV void load_a_frag(Frag& a, const unsigned short (*As)[LK], int rowBase, int lrow, int half) {
  const unsigned short* p = &As[rowBase + lrow][0];
  a.u[0] = *(const uint4*)(p + half * 8);
  a.u[1] = *(const uint4*)(p + 16 + half * 8);
}
DEV void load_b_frag(Frag& b, const unsigned short (*Bs)[LK], int rowBase, int lrow, int half) {
  const unsigned short* p = &Bs[rowBase + lrow][0];
  b.u[0] = *(const uint4*)(p + half * 8);
  b.u[1] = *(const uint4*)(p + 16 + half * 8);
}

// ---------------------------------------------------------------------------
// NT GEMM with bias:  Y[M,N] = A[M,K](bf16) * W[N,K]^T(bf16) + bias[N]
// 8 waves (4x2), wave tile 32x32 (2x2 WMMA), block tile 128x64.
// TDM double-buffered staging: wave0 issues stage k+1 while all compute k.
// ---------------------------------------------------------------------------
template<int OUT_F32>
__global__ __launch_bounds__(256)
void gemm_bias_nt(const unsigned short* __restrict__ A,
                  const unsigned short* __restrict__ W,
                  const float* __restrict__ bias,
                  unsigned short* __restrict__ obf,
                  float* __restrict__ of32,
                  int M, int N, int K) {
  __shared__ unsigned short As[2][BM][LK];
  __shared__ unsigned short Bs[2][BN][LK];

  const int tid  = threadIdx.x;
  const int lane = tid & 31;
  const int wave = tid >> 5;
  const int wrow = wave >> 1, wcol = wave & 1;
  const int half = lane >> 4, lrow = lane & 15;
  const int mBase = blockIdx.y * BM;
  const int nBase = blockIdx.x * BN;
  const int nsteps = K / BK;
  const bool ldr = is_wave0();                // scalar-branch TDM issuer

  v8f c[2][2];
  #pragma unroll
  for (int i = 0; i < 2; ++i)
    #pragma unroll
    for (int j = 0; j < 2; ++j)
      #pragma unroll
      for (int t = 0; t < 8; ++t) c[i][j][t] = 0.f;

  if (ldr) {
    tdm_load_2d_bf16((unsigned)(size_t)&As[0][0][0], A + (size_t)mBase * K, BM, BK, K);
    tdm_load_2d_bf16((unsigned)(size_t)&Bs[0][0][0], W + (size_t)nBase * K, BN, BK, K);
  }

  for (int k = 0; k < nsteps; ++k) {
    const int cur = k & 1, nxt = cur ^ 1;
    if (ldr) {
      if (k + 1 < nsteps) {
        const int k0 = (k + 1) * BK;
        tdm_load_2d_bf16((unsigned)(size_t)&As[nxt][0][0],
                         A + (size_t)mBase * K + k0, BM, BK, K);
        tdm_load_2d_bf16((unsigned)(size_t)&Bs[nxt][0][0],
                         W + (size_t)nBase * K + k0, BN, BK, K);
        __builtin_amdgcn_s_wait_tensorcnt(2);   // stage k complete (in-order)
      } else {
        __builtin_amdgcn_s_wait_tensorcnt(0);
      }
    }
    __syncthreads();                            // tile k visible to all waves

    Frag a[2], b[2];
    #pragma unroll
    for (int i = 0; i < 2; ++i) load_a_frag(a[i], As[cur], wrow * 32 + i * 16, lrow, half);
    #pragma unroll
    for (int j = 0; j < 2; ++j) load_b_frag(b[j], Bs[cur], wcol * 32 + j * 16, lrow, half);

    #pragma unroll
    for (int i = 0; i < 2; ++i)
      #pragma unroll
      for (int j = 0; j < 2; ++j)
        c[i][j] = wmma_bf16(a[i], b[j], c[i][j]);
    __syncthreads();                            // done reading buf before stage k+2 overwrite
  }

  #pragma unroll
  for (int i = 0; i < 2; ++i)
    #pragma unroll
    for (int j = 0; j < 2; ++j) {
      int n = nBase + wcol * 32 + j * 16 + lrow;
      float bv = bias[n];
      #pragma unroll
      for (int t = 0; t < 8; ++t) {
        int m = mBase + wrow * 32 + i * 16 + half * 8 + t;
        float val = c[i][j][t] + bv;
        if (OUT_F32) of32[(size_t)m * N + n] = val;
        else         obf[(size_t)m * N + n] = f2bf(val);
      }
    }
}

// ---------------------------------------------------------------------------
// Scores: logits[hz, m, n] = scale * Q[b,h,m,:] . K[b,g,n,:]  (masked -1e9)
// Strided views: Q lda=DM, K ldb=KVD. TDM handles the strides natively.
// Blocks fully above the causal diagonal exit early (softmax writes the zeros).
// ---------------------------------------------------------------------------
__global__ __launch_bounds__(256)
void attn_scores(const unsigned short* __restrict__ Q,
                 const unsigned short* __restrict__ Kb,
                 float* __restrict__ Sc,
                 const int* __restrict__ maskedp) {
  const int causal = maskedp[0];
  const int hz = blockIdx.z;                    // 0..B*NH-1
  const int b = hz >> 4, h = hz & 15, g = h >> 2;
  const int mBase = blockIdx.y * BM;
  const int nBase = blockIdx.x * BN;
  if (causal && nBase > mBase + BM - 1) return; // fully masked tile

  const unsigned short* qp = Q  + (size_t)b * SEQ * DM  + (size_t)h * HD;
  const unsigned short* kp = Kb + (size_t)b * SEQ * KVD + (size_t)g * HD;
  float* sp = Sc + (size_t)hz * SEQ * SEQ;

  __shared__ unsigned short As[2][BM][LK];
  __shared__ unsigned short Bs[2][BN][LK];

  const int tid  = threadIdx.x;
  const int lane = tid & 31;
  const int wave = tid >> 5;
  const int wrow = wave >> 1, wcol = wave & 1;
  const int half = lane >> 4, lrow = lane & 15;
  const int nsteps = HD / BK;                   // 2
  const bool ldr = is_wave0();

  v8f c[2][2];
  #pragma unroll
  for (int i = 0; i < 2; ++i)
    #pragma unroll
    for (int j = 0; j < 2; ++j)
      #pragma unroll
      for (int t = 0; t < 8; ++t) c[i][j][t] = 0.f;

  if (ldr) {
    tdm_load_2d_bf16((unsigned)(size_t)&As[0][0][0], qp + (size_t)mBase * DM,  BM, BK, DM);
    tdm_load_2d_bf16((unsigned)(size_t)&Bs[0][0][0], kp + (size_t)nBase * KVD, BN, BK, KVD);
  }

  #pragma unroll
  for (int k = 0; k < nsteps; ++k) {
    const int cur = k & 1, nxt = cur ^ 1;
    if (ldr) {
      if (k + 1 < nsteps) {
        const int k0 = (k + 1) * BK;
        tdm_load_2d_bf16((unsigned)(size_t)&As[nxt][0][0],
                         qp + (size_t)mBase * DM + k0, BM, BK, DM);
        tdm_load_2d_bf16((unsigned)(size_t)&Bs[nxt][0][0],
                         kp + (size_t)nBase * KVD + k0, BN, BK, KVD);
        __builtin_amdgcn_s_wait_tensorcnt(2);
      } else {
        __builtin_amdgcn_s_wait_tensorcnt(0);
      }
    }
    __syncthreads();

    Frag a[2], bfr[2];
    #pragma unroll
    for (int i = 0; i < 2; ++i) load_a_frag(a[i], As[cur], wrow * 32 + i * 16, lrow, half);
    #pragma unroll
    for (int j = 0; j < 2; ++j) load_b_frag(bfr[j], Bs[cur], wcol * 32 + j * 16, lrow, half);
    #pragma unroll
    for (int i = 0; i < 2; ++i)
      #pragma unroll
      for (int j = 0; j < 2; ++j)
        c[i][j] = wmma_bf16(a[i], bfr[j], c[i][j]);
    __syncthreads();
  }

  // streaming store: scores never fit in L2 (537 MB) -> non-temporal
  #pragma unroll
  for (int i = 0; i < 2; ++i)
    #pragma unroll
    for (int j = 0; j < 2; ++j) {
      int n = nBase + wcol * 32 + j * 16 + lrow;
      #pragma unroll
      for (int t = 0; t < 8; ++t) {
        int m = mBase + wrow * 32 + i * 16 + half * 8 + t;
        float val = c[i][j][t] * ATT_SCALE;
        if (causal && n > m) val = -1e9f;
        __builtin_nontemporal_store(val, sp + (size_t)m * SEQ + n);
      }
    }
}

// ---------------------------------------------------------------------------
// In-place row softmax over [B*H*S, S]; writes exact zeros above diagonal.
// ---------------------------------------------------------------------------
__global__ __launch_bounds__(256)
void softmax_rows(float* __restrict__ Sc, const int* __restrict__ maskedp) {
  const int causal = maskedp[0];
  const size_t rid = blockIdx.x;
  const int qi = (int)(rid & (SEQ - 1));
  float* row = Sc + rid * (size_t)SEQ;
  const int valid = causal ? (qi + 1) : SEQ;
  const int tid = threadIdx.x;
  __shared__ float red[256];

  float m = -3.0e38f;
  for (int j = tid; j < valid; j += 256) m = fmaxf(m, row[j]);
  red[tid] = m; __syncthreads();
  for (int s = 128; s > 0; s >>= 1) {
    if (tid < s) red[tid] = fmaxf(red[tid], red[tid + s]);
    __syncthreads();
  }
  m = red[0]; __syncthreads();

  float sum = 0.f;
  for (int j = tid; j < valid; j += 256) {
    float e = __expf(row[j] - m);
    row[j] = e;
    sum += e;
  }
  red[tid] = sum; __syncthreads();
  for (int s = 128; s > 0; s >>= 1) {
    if (tid < s) red[tid] += red[tid + s];
    __syncthreads();
  }
  const float inv = 1.f / red[0];

  for (int j = tid; j < SEQ; j += 256) {
    float p = (j < valid) ? row[j] * inv : 0.f;
    __builtin_nontemporal_store(p, row + j);    // read once by PV, then dead
  }
}

// ---------------------------------------------------------------------------
// PV: out[b, m, h*HD + d] = sum_k P[hz, m, k] * V[b, k, g*HD + d]
// P read fp32 (non-temporal) from d_out post-softmax, converted to bf16
// while staging; V staged transposed into LDS. Causal: K-loop stops at the
// diagonal block (P is exactly zero above it).
// ---------------------------------------------------------------------------
__global__ __launch_bounds__(256)
void attn_av(const float* __restrict__ P,
             const unsigned short* __restrict__ V,
             unsigned short* __restrict__ Ao,
             const int* __restrict__ maskedp) {
  const int causal = maskedp[0];
  const int hz = blockIdx.z;
  const int b = hz >> 4, h = hz & 15, g = h >> 2;
  const int mBase = blockIdx.y * BM;
  const float* pp = P + (size_t)hz * SEQ * SEQ;
  const unsigned short* vp = V + (size_t)b * SEQ * KVD + (size_t)g * HD;
  const int kEnd = causal ? (mBase + BM) : SEQ;

  __shared__ unsigned short As[BM][LK];
  __shared__ unsigned short Bs[BN][LK];

  const int tid  = threadIdx.x;
  const int lane = tid & 31, wave = tid >> 5;
  const int wrow = wave >> 1, wcol = wave & 1;
  const int half = lane >> 4, lrow = lane & 15;

  v8f c[2][2];
  #pragma unroll
  for (int i = 0; i < 2; ++i)
    #pragma unroll
    for (int j = 0; j < 2; ++j)
      #pragma unroll
      for (int t = 0; t < 8; ++t) c[i][j][t] = 0.f;

  for (int k0 = 0; k0 < kEnd; k0 += BK) {
    // A tile: 128x32 probs fp32 -> bf16 (1024 float4 chunks / 256 threads)
    #pragma unroll
    for (int it = 0; it < 4; ++it) {
      int idx = tid + it * 256;
      int r = idx >> 3, c4 = idx & 7;
      const float* src = pp + (size_t)(mBase + r) * SEQ + k0 + c4 * 4;
      f32x4 d = __builtin_nontemporal_load((const f32x4*)src);
      if (k0 + BK < kEnd) __builtin_prefetch(src + BK, 0, 0);  // next k-step tile
      uint2 pk;
      pk.x = (unsigned)f2bf(d[0]) | ((unsigned)f2bf(d[1]) << 16);
      pk.y = (unsigned)f2bf(d[2]) | ((unsigned)f2bf(d[3]) << 16);
      *(uint2*)(&As[r][c4 * 4]) = pk;
    }
    // B tile: V^T — Bs[n][k] = V[k0+k][n], 32(k) x 64(n) bf16
    {
      int k = tid >> 3, n8 = (tid & 7) * 8;
      union { uint4 u; unsigned short s[8]; } d;
      d.u = *(const uint4*)(vp + (size_t)(k0 + k) * KVD + n8);
      #pragma unroll
      for (int e = 0; e < 8; ++e) Bs[n8 + e][k] = d.s[e];
    }
    __syncthreads();

    Frag a[2], bfr[2];
    #pragma unroll
    for (int i = 0; i < 2; ++i) load_a_frag(a[i], As, wrow * 32 + i * 16, lrow, half);
    #pragma unroll
    for (int j = 0; j < 2; ++j) load_b_frag(bfr[j], Bs, wcol * 32 + j * 16, lrow, half);
    #pragma unroll
    for (int i = 0; i < 2; ++i)
      #pragma unroll
      for (int j = 0; j < 2; ++j)
        c[i][j] = wmma_bf16(a[i], bfr[j], c[i][j]);
    __syncthreads();
  }

  #pragma unroll
  for (int i = 0; i < 2; ++i)
    #pragma unroll
    for (int j = 0; j < 2; ++j) {
      int dcol = wcol * 32 + j * 16 + lrow;     // 0..63
      #pragma unroll
      for (int t = 0; t < 8; ++t) {
        int m = mBase + wrow * 32 + i * 16 + half * 8 + t;
        Ao[((size_t)b * SEQ + m) * DM + h * HD + dcol] = f2bf(c[i][j][t]);
      }
    }
}

// ---------------------------------------------------------------------------
// host launcher
// ---------------------------------------------------------------------------
extern "C" void kernel_launch(void* const* d_in, const int* in_sizes, int n_in,
                              void* d_out, int out_size, void* d_ws, size_t ws_size,
                              hipStream_t stream) {
  (void)in_sizes; (void)n_in; (void)out_size; (void)ws_size;

  const float* x  = (const float*)d_in[0];
  const float* Wq = (const float*)d_in[1];
  const float* bq = (const float*)d_in[2];
  const float* Wk = (const float*)d_in[3];
  const float* bk = (const float*)d_in[4];
  const float* Wv = (const float*)d_in[5];
  const float* bv = (const float*)d_in[6];
  const float* Wo = (const float*)d_in[7];
  const float* bo = (const float*)d_in[8];
  const int* masked = (const int*)d_in[9];

  float* out    = (float*)d_out;                    // [B,S,D]
  float* scores = out + (size_t)BB * SEQ * DM;      // [B,H,S,S]

  unsigned short* w = (unsigned short*)d_ws;        // ~33 MB bf16 scratch
  size_t off = 0;
  unsigned short* xbf = w + off; off += (size_t)BB * SEQ * DM;
  unsigned short* wqb = w + off; off += (size_t)DM * DM;
  unsigned short* wkb = w + off; off += (size_t)KVD * DM;
  unsigned short* wvb = w + off; off += (size_t)KVD * DM;
  unsigned short* wob = w + off; off += (size_t)DM * DM;
  unsigned short* qbf = w + off; off += (size_t)BB * SEQ * DM;
  unsigned short* kbf = w + off; off += (size_t)BB * SEQ * KVD;
  unsigned short* vbf = w + off; off += (size_t)BB * SEQ * KVD;
  unsigned short* abf = w + off; off += (size_t)BB * SEQ * DM;

  const dim3 blk(256);
  auto cast = [&](const float* in, unsigned short* o, int n) {
    int n4 = n / 4;
    cast_f32_bf16<<<dim3((n4 + 255) / 256), blk, 0, stream>>>(in, o, n4);
  };

  // 1) bf16 staging
  cast(x,  xbf, BB * SEQ * DM);
  cast(Wq, wqb, DM * DM);
  cast(Wk, wkb, KVD * DM);
  cast(Wv, wvb, KVD * DM);
  cast(Wo, wob, DM * DM);

  // 2) projections (bf16 out)
  gemm_bias_nt<0><<<dim3(DM  / BN, BB * SEQ / BM), blk, 0, stream>>>(
      xbf, wqb, bq, qbf, nullptr, BB * SEQ, DM,  DM);
  gemm_bias_nt<0><<<dim3(KVD / BN, BB * SEQ / BM), blk, 0, stream>>>(
      xbf, wkb, bk, kbf, nullptr, BB * SEQ, KVD, DM);
  gemm_bias_nt<0><<<dim3(KVD / BN, BB * SEQ / BM), blk, 0, stream>>>(
      xbf, wvb, bv, vbf, nullptr, BB * SEQ, KVD, DM);

  // 3) masked logits -> 4) softmax (in place, fills zeros above diagonal)
  attn_scores<<<dim3(SEQ / BN, SEQ / BM, BB * NH), blk, 0, stream>>>(
      qbf, kbf, scores, masked);
  softmax_rows<<<dim3(BB * NH * SEQ), blk, 0, stream>>>(scores, masked);

  // 5) PV into [B,S,D] bf16 layout
  attn_av<<<dim3(1, SEQ / BM, BB * NH), blk, 0, stream>>>(
      scores, vbf, abf, masked);

  // 6) O projection (fp32 straight into d_out)
  gemm_bias_nt<1><<<dim3(DM / BN, BB * SEQ / BM), blk, 0, stream>>>(
      abf, wob, bo, nullptr, out, BB * SEQ, DM, DM);
}
